// PinnLoss_13804024889406
// MI455X (gfx1250) — matrix operand order repaired and checked by reading, not compiled, not run on hardware
//
#include <hip/hip_runtime.h>
#include <hip/hip_bf16.h>

typedef __attribute__((ext_vector_type(16))) _Float16 v16h;
typedef __attribute__((ext_vector_type(8)))  _Float16 v8h;
typedef __attribute__((ext_vector_type(8)))  float    v8f;

#define C_FD 0.01f
#define NW   8      // waves per workgroup (256 threads)
#define HID  264    // padded LDS row stride in halfs (256 + 8) -> 528B, conflict-free

// tanh(x) = 1 - 2/(exp(2x)+1); exp via hardware exp2, rcp via v_rcp_f32.
// Saturates correctly to +-1 for large |x|.
__device__ __forceinline__ float fast_tanh(float x) {
  float e = __builtin_amdgcn_exp2f(x * 2.8853900817779268f); // 2*log2(e)
  return 1.0f - 2.0f * __builtin_amdgcn_rcpf(e + 1.0f);
}

// Load 16 consecutive halfs as two 16B LDS loads (ds_load_b128 x2).
__device__ __forceinline__ v16h ld_frag(const _Float16* p) {
  v8h lo = *(const v8h*)p;
  v8h hi = *(const v8h*)(p + 8);
  v16h r;
#pragma unroll
  for (int q = 0; q < 8; ++q) { r[q] = lo[q]; r[8 + q] = hi[q]; }
  return r;
}

__device__ __forceinline__ v8f wmma_f16(v16h a, v16h b, v8f c) {
  return __builtin_amdgcn_wmma_f32_16x16x32_f16(false, a, false, b, (short)0, c,
                                                false, false);
}

__global__ void zero_out_kernel(float* out) { out[0] = 0.0f; }

__global__ void mse_kernel(const float* __restrict__ p, const float* __restrict__ t,
                           float* __restrict__ out, int n, float invM) {
  __shared__ float red[256];
  float s = 0.0f;
  for (int i = blockIdx.x * blockDim.x + threadIdx.x; i < n; i += gridDim.x * blockDim.x) {
    float d = p[i] - t[i];
    s += d * d;
  }
  red[threadIdx.x] = s;
  __syncthreads();
  for (int off = 128; off > 0; off >>= 1) {
    if (threadIdx.x < off) red[threadIdx.x] += red[threadIdx.x + off];
    __syncthreads();
  }
  if (threadIdx.x == 0) atomicAdd(out, red[0] * invM);
}

// Fused PINN kernel: per wave, 16 points; forward (x->h1->h2->V) and VJP back to
// g_x via WMMA GEMMs against LDS-resident W2 / W2^T, for the base points and the
// x0-shifted points (finite difference), then the Black-Scholes PDE residual.
__global__ __launch_bounds__(256, 1)
void pinn_kernel(const float* __restrict__ dom,
                 const float* __restrict__ W1, const float* __restrict__ b1,
                 const float* __restrict__ W2, const float* __restrict__ b2,
                 const float* __restrict__ W3, const float* __restrict__ b3,
                 float* __restrict__ out, int numGroups, float invM) {
  extern __shared__ char smem[];
  _Float16* w2f = (_Float16*)smem;           // [256][HID] W2 row-major  (backward B frags: B=W2^T)
  _Float16* w2t = w2f + 256 * HID;           // [256][HID] W2^T row-major (forward  B frags: B=W2)
  float* w1s   = (float*)(w2t + 256 * HID);  // [5][256]
  float* b1s   = w1s + 5 * 256;              // [256]
  float* b2s   = b1s + 256;                  // [256]
  float* w3s   = b2s + 256;                  // [256]
  float* xt_all = w3s + 256;                 // [NW][16][8] per-wave x tiles
  _Float16* pb_all = (_Float16*)(xt_all + NW * 16 * 8); // [NW][16][40] gz2 pair buffers

  const int tid = threadIdx.x;

  // ---- stage weights to LDS (f16 W2 twice, padded) ----
  for (int idx = tid; idx < 256 * 256; idx += 256) {
    int i = idx >> 8, j = idx & 255;
    _Float16 h = (_Float16)W2[idx];
    w2f[i * HID + j] = h;   // W2[i][j]
    w2t[j * HID + i] = h;   // W2^T[j][i]
  }
  for (int idx = tid; idx < 5 * 256; idx += 256) w1s[idx] = W1[idx];
  if (tid < 256) { b1s[tid] = b1[tid]; b2s[tid] = b2[tid]; w3s[tid] = W3[tid]; }
  __syncthreads();

  const float b3v = b3[0];
  const int wave = tid >> 5;
  const int lane = tid & 31;
  const int lm = lane & 15;   // column / M-row selector within fragments
  const int lh = lane >> 4;   // half-wave selector
  float*    xt = xt_all + wave * (16 * 8);
  _Float16* pb = pb_all + wave * (16 * 40);

  float total = 0.0f;

  for (int g = blockIdx.x; g < numGroups; g += gridDim.x) {
    const int base = (g * NW + wave) * 16;   // first point of this wave's tile

    if (lane < 16) {
      const float* xp = dom + (long)(base + lane) * 5;
#pragma unroll
      for (int d = 0; d < 5; ++d) xt[lane * 8 + d] = xp[d];
      // prefetch next group's points into cache (global_prefetch_b8)
      const int gn = g + gridDim.x;
      if (gn < numGroups)
        __builtin_prefetch(dom + (long)((gn * NW + wave) * 16 + lane) * 5, 0, 1);
    }
    float xa[5];
#pragma unroll
    for (int d = 0; d < 5; ++d) xa[d] = xt[lm * 8 + d];

    float res_v[8], res_s[8], res_t[8], res_sb[8];

#pragma unroll 1
    for (int pass = 0; pass < 2; ++pass) {
      const float fd = pass ? C_FD : 0.0f;
      const float x0 = xa[0] + fd;

      // ---- layer 1 forward: H1 A-fragments (16 pts x 256), f16 A layout.
      // lane holds row M=lm; halfs[8p+j]=K(32c+16*lh+8p+j) (contiguous k order, matches B frags).
      v16h afr[8];
#pragma unroll
      for (int c = 0; c < 8; ++c) {
#pragma unroll
        for (int p = 0; p < 2; ++p) {
#pragma unroll
          for (int j = 0; j < 8; ++j) {
            const int k = 32 * c + 16 * lh + 8 * p + j;
            float z = b1s[k] + x0 * w1s[k] + xa[1] * w1s[256 + k] + xa[2] * w1s[512 + k]
                             + xa[3] * w1s[768 + k] + xa[4] * w1s[1024 + k];
            afr[c][p * 8 + j] = (_Float16)fast_tanh(z);
          }
        }
      }

      const v8f vz = {0.f, 0.f, 0.f, 0.f, 0.f, 0.f, 0.f, 0.f};
      float vacc[8];
#pragma unroll
      for (int r = 0; r < 8; ++r) vacc[r] = 0.0f;

      // ---- layer 2 forward: Z2 = H1 @ W2, per 32-wide column pair; the
      // resulting gz2 chunk is converted through the LDS pair buffer into
      // A-fragment layout and kept RESIDENT in registers (gafr, f16: only
      // 64 VGPRs for the whole 16x256 Gz2 tile).
      v16h gafr[8];
#pragma unroll
      for (int pr = 0; pr < 8; ++pr) {
        // two independent accumulator chains (t=0,1), B frags double-buffered
        const _Float16* brow = w2t + (pr * 32 + lm) * HID + 16 * lh;
        v8f accs[2] = {vz, vz};
        v16h fcur0 = ld_frag(brow);
        v16h fcur1 = ld_frag(brow + 16 * HID);
#pragma unroll
        for (int kc = 0; kc < 8; ++kc) {
          v16h fnx0 = fcur0, fnx1 = fcur1;
          if (kc < 7) {
            fnx0 = ld_frag(brow + (kc + 1) * 32);
            fnx1 = ld_frag(brow + 16 * HID + (kc + 1) * 32);
          }
          accs[0] = wmma_f16(afr[kc], fcur0, accs[0]);
          accs[1] = wmma_f16(afr[kc], fcur1, accs[1]);
          fcur0 = fnx0;
          fcur1 = fnx1;
        }
        // elementwise: h2 = tanh(z2+b2), V += h2*W3, gz2 = (1-h2^2)*W3
#pragma unroll
        for (int t = 0; t < 2; ++t) {
          const int n = pr * 32 + t * 16 + lm;
          const float b2n = b2s[n];
          const float w3n = w3s[n];
#pragma unroll
          for (int r = 0; r < 8; ++r) {
            float h2 = fast_tanh(accs[t][r] + b2n);   // C/D layout: M=r+8*lh, N=lm
            vacc[r] += h2 * w3n;
            float gz2 = (1.0f - h2 * h2) * w3n;
            pb[(r + 8 * lh) * 40 + t * 16 + lm] = (_Float16)gz2;
          }
        }
        // re-load this 32-wide chunk of Gz2 in A-fragment layout
        v8h glo = *(const v8h*)(pb + lm * 40 + 16 * lh);
        v8h ghi = *(const v8h*)(pb + lm * 40 + 16 * lh + 8);
        v16h ga;
#pragma unroll
        for (int q = 0; q < 8; ++q) { ga[q] = glo[q]; ga[8 + q] = ghi[q]; }
        gafr[pr] = ga;
      }

      // ---- layer 2 backward (Gh1 = Gz2 @ W2^T) fused with layer 1 backward.
      // Two i-tiles at a time (two independent WMMA chains), results consumed
      // immediately so only 16 accumulator VGPRs are ever live.
      float xr[8][5];
#pragma unroll
      for (int r = 0; r < 8; ++r) {
#pragma unroll
        for (int d = 0; d < 5; ++d) xr[r][d] = xt[(r + 8 * lh) * 8 + d];
        xr[r][0] += fd;
      }
      float as[8], at[8];
#pragma unroll
      for (int r = 0; r < 8; ++r) { as[r] = 0.0f; at[r] = 0.0f; }

#pragma unroll 1
      for (int it2 = 0; it2 < 8; ++it2) {
        const _Float16* r0 = w2f + (it2 * 32 + lm) * HID + 16 * lh;  // i-tile 2*it2
        const _Float16* r1 = r0 + 16 * HID;                          // i-tile 2*it2+1
        v8f ac0 = vz, ac1 = vz;
        v16h c0 = ld_frag(r0);
        v16h c1 = ld_frag(r1);
#pragma unroll
        for (int pr = 0; pr < 8; ++pr) {
          v16h n0 = c0, n1 = c1;
          if (pr < 7) {
            n0 = ld_frag(r0 + (pr + 1) * 32);
            n1 = ld_frag(r1 + (pr + 1) * 32);
          }
          ac0 = wmma_f16(gafr[pr], c0, ac0);
          ac1 = wmma_f16(gafr[pr], c1, ac1);
          c0 = n0;
          c1 = n1;
        }
        // gz1 = (1-h1^2)*gh1, fold into g_x[0] (V_s) and g_x[3] (V_t)
#pragma unroll
        for (int t = 0; t < 2; ++t) {
          const int i = (it2 * 2 + t) * 16 + lm;
          const float w10 = w1s[i], w11 = w1s[256 + i], w12 = w1s[512 + i];
          const float w13 = w1s[768 + i], w14 = w1s[1024 + i], b1i = b1s[i];
#pragma unroll
          for (int r = 0; r < 8; ++r) {
            float z = b1i + xr[r][0] * w10 + xr[r][1] * w11 + xr[r][2] * w12
                          + xr[r][3] * w13 + xr[r][4] * w14;
            float h1v = fast_tanh(z);
            float gh1 = t ? ac1[r] : ac0[r];          // C/D layout: M=r+8*lh, N=lm
            float gz1 = (1.0f - h1v * h1v) * gh1;
            as[r] += gz1 * w10;
            at[r] += gz1 * w13;
          }
        }
      }

      // ---- reduce across the 16 lanes of each half-wave
#pragma unroll
      for (int r = 0; r < 8; ++r) {
        float s = as[r], t2 = at[r], v = vacc[r];
#pragma unroll
        for (int m = 1; m <= 8; m <<= 1) {
          s  += __shfl_xor(s, m, 32);
          t2 += __shfl_xor(t2, m, 32);
          v  += __shfl_xor(v, m, 32);
        }
        if (pass == 0) { res_s[r] = s; res_t[r] = t2; res_v[r] = v; }
        else           { res_sb[r] = s; }
      }
    } // pass

    // ---- PDE residual per point (lanes 0 and 16 hold the sums)
    if (lm == 0) {
#pragma unroll
      for (int r = 0; r < 8; ++r) {
        const int m = r + 8 * lh;
        const float S  = xt[m * 8 + 0];
        const float sg = xt[m * 8 + 2];
        const float rr = xt[m * 8 + 4];
        const float Vs  = res_s[r];
        const float Vss = (res_sb[r] - Vs) * (1.0f / C_FD);
        const float Va  = res_v[r] + b3v;
        const float err = res_t[r] + 0.5f * sg * sg * S * S * Vss + rr * S * Vs - rr * Va;
        total += err * err;
      }
    }
  } // group loop

  if (lm == 0) atomicAdd(out, total * invM);
}

extern "C" void kernel_launch(void* const* d_in, const int* in_sizes, int n_in,
                              void* d_out, int out_size, void* d_ws, size_t ws_size,
                              hipStream_t stream) {
  (void)n_in; (void)out_size; (void)d_ws; (void)ws_size;
  const float* pred = (const float*)d_in[0];
  const float* targ = (const float*)d_in[1];
  const float* dom  = (const float*)d_in[2];
  const float* W1   = (const float*)d_in[3];
  const float* b1   = (const float*)d_in[4];
  const float* W2   = (const float*)d_in[5];
  const float* b2   = (const float*)d_in[6];
  const float* W3   = (const float*)d_in[7];
  const float* b3   = (const float*)d_in[8];
  float* out = (float*)d_out;

  const int BN = in_sizes[0];               // B*N points (262144)
  const int numGroups = BN / (16 * NW);     // 128 points per workgroup-iteration
  const float invM = 1.0f / (float)BN;

  zero_out_kernel<<<1, 1, 0, stream>>>(out);
  mse_kernel<<<256, 256, 0, stream>>>(pred, targ, out, BN, invM);

  const size_t smem = (size_t)(2 * 256 * HID) * sizeof(_Float16)         // W2 + W2^T
                    + (size_t)(5 * 256 + 3 * 256 + NW * 16 * 8) * sizeof(float)
                    + (size_t)(NW * 16 * 40) * sizeof(_Float16);
  int grid = numGroups < 1024 ? numGroups : 1024;
  pinn_kernel<<<grid, 256, smem, stream>>>(dom, W1, b1, W2, b2, W3, b3, out,
                                           numGroups, invM);
}
